// ConceptPrototypeNetwork_23905787969834
// MI455X (gfx1250) — matrix operand order repaired and checked by reading, not compiled, not run on hardware
//
#include <hip/hip_runtime.h>
#include <hip/hip_bf16.h>

// ---- problem constants (match reference) ----
#define BATCH 16384
#define NCON  4096
#define DIM   64
#define EPSV  1e-7f
#define LN2F  0.6931471805599453f

typedef __attribute__((ext_vector_type(16))) _Float16 v16h;
typedef __attribute__((ext_vector_type(8)))  float    v8f;
typedef __attribute__((ext_vector_type(4)))  _Float16 h4;

union FragU { v16h h; uint4 q[2]; };

// ---------------------------------------------------------------------------
// Pre-pass: fp32 -> fp16 conversion + per-row squared norm.
// 16 lanes per row, float4 per lane (coalesced), shuffle-reduce within 16.
// ---------------------------------------------------------------------------
__global__ __launch_bounds__(256) void cpn_convert_norm(
    const float* __restrict__ src, _Float16* __restrict__ dst,
    float* __restrict__ norms, int rows) {
  int t = blockIdx.x * blockDim.x + threadIdx.x;
  int r = t >> 4;
  int c = (t & 15) << 2;
  if (r >= rows) return;
  float4 v = *(const float4*)(src + (size_t)r * DIM + c);
  float s = v.x * v.x + v.y * v.y + v.z * v.z + v.w * v.w;
  s += __shfl_xor(s, 1, 16);
  s += __shfl_xor(s, 2, 16);
  s += __shfl_xor(s, 4, 16);
  s += __shfl_xor(s, 8, 16);
  h4 hv;
  hv.x = (_Float16)v.x; hv.y = (_Float16)v.y;
  hv.z = (_Float16)v.z; hv.w = (_Float16)v.w;
  *(h4*)(dst + (size_t)r * DIM + c) = hv;
  if ((t & 15) == 0) norms[r] = s;
}

// ---------------------------------------------------------------------------
// Main kernel: 128x128 output tile per 256-thread block (8 wave32 waves).
// Prototype tile staged in LDS (padded pitch), WMMA f32_16x16x32_f16,
// hardware-transcendental Poincare-distance epilogue fused.
// ---------------------------------------------------------------------------
#define TILE_M 128
#define TILE_N 128
#define LDS_PITCH 72  // halves per row: 64 data + 8 pad (144 B, 16B aligned)

__global__ __launch_bounds__(256) void cpn_poincare_gemm(
    const _Float16* __restrict__ Ah, const _Float16* __restrict__ Ph,
    const float* __restrict__ x2n, const float* __restrict__ p2n,
    float* __restrict__ out) {
  __shared__ _Float16 Bs[TILE_N][LDS_PITCH];

  const int tid   = threadIdx.x;
  const int lane  = tid & 31;
  const int wave  = tid >> 5;
  const int hi    = lane >> 4;   // half-wave selector
  const int l16   = lane & 15;
  const int tileN = blockIdx.x * TILE_N;
  const int tileM = blockIdx.y * TILE_M;

  // ---- stage 128x64 f16 prototype tile into LDS: 1024 x 16B chunks ----
#pragma unroll
  for (int i = 0; i < 4; ++i) {
    int chunk = tid + i * 256;
    int row   = chunk >> 3;
    int col   = (chunk & 7) << 3;  // in halves
    *(uint4*)&Bs[row][col] =
        *(const uint4*)(Ph + (size_t)(tileN + row) * DIM + col);
  }
  __syncthreads();

  const int mBase = tileM + wave * 16;

  v8f acc[8];
  const v8f vzero = {0.f, 0.f, 0.f, 0.f, 0.f, 0.f, 0.f, 0.f};
#pragma unroll
  for (int nb = 0; nb < 8; ++nb) acc[nb] = vzero;

  // ---- K=64 as two 16x16x32 WMMA steps ----
#pragma unroll
  for (int ks = 0; ks < 2; ++ks) {
    const int k0 = ks * 32;
    // A fragment: lane l holds row (mBase+l16), K = k0 + hi*8 + {0..7, 16..23}
    FragU a;
    const _Float16* ap = Ah + (size_t)(mBase + l16) * DIM + k0 + hi * 8;
    a.q[0] = *(const uint4*)(ap);
    a.q[1] = *(const uint4*)(ap + 16);
#pragma unroll
    for (int nb = 0; nb < 8; ++nb) {
      // B fragment: lane l holds col (tileN+nb*16+l16), K = k0 + hi*8 + {0..7, 16..23}
      FragU b;
      const _Float16* bp = &Bs[nb * 16 + l16][k0 + hi * 8];
      b.q[0] = *(const uint4*)(bp);
      b.q[1] = *(const uint4*)(bp + 16);
      acc[nb] = __builtin_amdgcn_wmma_f32_16x16x32_f16(
          false, a.h, false, b.h, (short)0, acc[nb], false, false);
    }
  }

  // ---- fused Poincare-distance epilogue (hardware trans ops) ----
  // C/D layout: VGPR j -> row mBase + hi*8 + j, col = tileN + nb*16 + l16.
  // d = acosh(1 + t), t = 2*sq/denom  ->  d = ln(1 + t + sqrt(t*(t+2)))
  float xv[8], rx[8];  // x2 and 2/(1 - x2) per row
  {
    const float* xp = x2n + mBase + hi * 8;
    float4 xa = *(const float4*)(xp);
    float4 xb = *(const float4*)(xp + 4);
    xv[0] = xa.x; xv[1] = xa.y; xv[2] = xa.z; xv[3] = xa.w;
    xv[4] = xb.x; xv[5] = xb.y; xv[6] = xb.z; xv[7] = xb.w;
#pragma unroll
    for (int j = 0; j < 8; ++j)
      rx[j] = 2.0f * __builtin_amdgcn_rcpf(fmaxf(1.0f - xv[j], EPSV));
  }

  float p2v[8], rp[8];  // p2 and 1/(1 - p2) per N-block
#pragma unroll
  for (int nb = 0; nb < 8; ++nb) {
    p2v[nb] = p2n[tileN + nb * 16 + l16];
    rp[nb]  = __builtin_amdgcn_rcpf(fmaxf(1.0f - p2v[nb], EPSV));
  }

#pragma unroll
  for (int j = 0; j < 8; ++j) {
    float* __restrict__ rowp =
        out + (size_t)(mBase + hi * 8 + j) * NCON + tileN + l16;
#pragma unroll
    for (int nb = 0; nb < 8; ++nb) {
      float xp = acc[nb][j];
      float sq = fmaxf(__builtin_fmaf(-2.0f, xp, xv[j] + p2v[nb]), 0.0f);
      float t  = fmaxf(sq * rx[j] * rp[nb], EPSV);
      float s  = __builtin_amdgcn_sqrtf(__builtin_fmaf(t, t, t + t));
      float d  = __builtin_amdgcn_logf(1.0f + t + s) * LN2F;
      rowp[nb * 16] = d;
    }
  }
}

// ---------------------------------------------------------------------------
// Host-side launch (graph-capture safe: only kernel launches on `stream`).
// ---------------------------------------------------------------------------
extern "C" void kernel_launch(void* const* d_in, const int* in_sizes, int n_in,
                              void* d_out, int out_size, void* d_ws, size_t ws_size,
                              hipStream_t stream) {
  const float* emb  = (const float*)d_in[0];  // (16384, 64) f32
  const float* prot = (const float*)d_in[1];  // (4096, 64)  f32
  float* out = (float*)d_out;                 // (16384, 4096) f32

  // workspace layout
  char* ws = (char*)d_ws;
  _Float16* Ah = (_Float16*)(ws);                                  // 2 MB
  _Float16* Ph = (_Float16*)(ws + (size_t)BATCH * DIM * 2);        // 512 KB
  float*    x2 = (float*)(ws + (size_t)BATCH * DIM * 2
                             + (size_t)NCON * DIM * 2);            // 64 KB
  float*    p2 = (float*)(ws + (size_t)BATCH * DIM * 2
                             + (size_t)NCON * DIM * 2
                             + (size_t)BATCH * 4);                 // 16 KB

  // pre-pass: convert + row norms
  {
    int threadsA = BATCH * 16;
    cpn_convert_norm<<<(threadsA + 255) / 256, 256, 0, stream>>>(emb, Ah, x2, BATCH);
    int threadsP = NCON * 16;
    cpn_convert_norm<<<(threadsP + 255) / 256, 256, 0, stream>>>(prot, Ph, p2, NCON);
  }

  // main fused GEMM + hyperbolic-distance epilogue
  dim3 grid(NCON / TILE_N, BATCH / TILE_M);  // (32, 128)
  cpn_poincare_gemm<<<grid, 256, 0, stream>>>(Ah, Ph, x2, p2, out);
}
